// CausalSelfAttention_30339648979677
// MI455X (gfx1250) — matrix operand order repaired and checked by reading, not compiled, not run on hardware
//
#include <hip/hip_runtime.h>

typedef __attribute__((ext_vector_type(16))) _Float16 v16h;
typedef __attribute__((ext_vector_type(8)))  _Float16 h8;
typedef __attribute__((ext_vector_type(4)))  _Float16 h4;
typedef __attribute__((ext_vector_type(8)))  float    v8f;
typedef __attribute__((ext_vector_type(4)))  unsigned u32x4;
typedef __attribute__((ext_vector_type(8)))  int      i32x8;
typedef __attribute__((ext_vector_type(4)))  int      i32x4;

union FragA { v16h v; h8 h[2]; };

#define BB 8
#define SS 1024
#define DD 1024
#define HH 16

__device__ inline unsigned lds_off(const void* p) {
  // generic pointer to __shared__: low 32 bits are the LDS byte offset
  return (unsigned)(unsigned long long)p;
}

// XOR-butterfly reductions within 16-lane groups via ds_swizzle immediates
// (group-of-32 mode: offset = xor_mask<<10 | and_mask 0x1f; xor<16 keeps the
// exchange inside each half-wave group).
__device__ inline float red_max16(float x) {
  x = fmaxf(x, __int_as_float(__builtin_amdgcn_ds_swizzle(__float_as_int(x), 0x041f)));
  x = fmaxf(x, __int_as_float(__builtin_amdgcn_ds_swizzle(__float_as_int(x), 0x081f)));
  x = fmaxf(x, __int_as_float(__builtin_amdgcn_ds_swizzle(__float_as_int(x), 0x101f)));
  x = fmaxf(x, __int_as_float(__builtin_amdgcn_ds_swizzle(__float_as_int(x), 0x201f)));
  return x;
}
__device__ inline float red_sum16(float x) {
  x += __int_as_float(__builtin_amdgcn_ds_swizzle(__float_as_int(x), 0x041f));
  x += __int_as_float(__builtin_amdgcn_ds_swizzle(__float_as_int(x), 0x081f));
  x += __int_as_float(__builtin_amdgcn_ds_swizzle(__float_as_int(x), 0x101f));
  x += __int_as_float(__builtin_amdgcn_ds_swizzle(__float_as_int(x), 0x201f));
  return x;
}

// ---------------------------------------------------------------------------
// Tensor Data Mover: DMA a 2D f16 tile [tile_rows x tile_cols] (row-major,
// row stride = row_stride elements) from global to LDS, with hardware row
// padding (pad_iv: interval = 2^(iv+1) DWORDs; pad_amt: (amt+1) DWORDs).
// D# packing per CDNA5 ISA.
// ---------------------------------------------------------------------------
__device__ inline void tdm_load_tile_f16(
    const _Float16* gptr, unsigned lds_byte,
    unsigned tile_cols, unsigned tile_rows,
    unsigned long long row_stride, unsigned pad_iv, unsigned pad_amt)
{
  const unsigned long long ga = (unsigned long long)gptr;
  u32x4 g0;
  g0[0] = 1u;                                             // count=1, no gather
  g0[1] = lds_byte;                                       // lds_addr
  g0[2] = (unsigned)(ga & 0xffffffffu);                   // global_addr[31:0]
  g0[3] = (unsigned)((ga >> 32) & 0x1ffffffu) | (2u << 30); // [56:32] | type=2

  const unsigned td0 = tile_cols;   // tensor_dim0 (tile == tensor: never OOB)
  const unsigned td1 = tile_rows;
  i32x8 g1;
  g1[0] = (int)((1u << 16) |        // data_size = 2 bytes
                (1u << 20) |        // pad_enable
                (pad_iv  << 22) |
                (pad_amt << 25));
  g1[1] = (int)((td0 & 0xffffu) << 16);                     // tensor_dim0[15:0]
  g1[2] = (int)(((td0 >> 16) & 0xffffu) | ((td1 & 0xffffu) << 16));
  g1[3] = (int)(((td1 >> 16) & 0xffffu) | ((tile_cols & 0xffffu) << 16)); // tile_dim0
  g1[4] = (int)(tile_rows & 0xffffu);                       // tile_dim1, tile_dim2=0
  g1[5] = (int)(row_stride & 0xffffffffu);                  // tensor_dim0_stride
  g1[6] = (int)((row_stride >> 32) & 0xffffu);
  g1[7] = 0;

  const i32x4 z4 = {0, 0, 0, 0};
#if defined(__clang_major__) && (__clang_major__ >= 23)
  const i32x8 z8 = {0, 0, 0, 0, 0, 0, 0, 0};
  __builtin_amdgcn_tensor_load_to_lds(g0, g1, z4, z4, z8, 0);
#else
  __builtin_amdgcn_tensor_load_to_lds(g0, g1, z4, z4, 0);
#endif
}

// ---------------------------------------------------------------------------
// Prepass: f32 -> f16 (4 elements / thread)
// ---------------------------------------------------------------------------
__global__ __launch_bounds__(256) void f32_to_f16_vec(
    const float* __restrict__ in, _Float16* __restrict__ out)
{
  const size_t i = ((size_t)blockIdx.x * 256 + threadIdx.x) * 4;
  const float4 f = *(const float4*)(in + i);
  h4 hh; hh[0] = (_Float16)f.x; hh[1] = (_Float16)f.y;
         hh[2] = (_Float16)f.z; hh[3] = (_Float16)f.w;
  *(h4*)(out + i) = hh;
}

// ---------------------------------------------------------------------------
// Prepass: W[K][N] f32 -> WT[N][K] f16, 32x32 LDS-tiled transpose.
// ---------------------------------------------------------------------------
__global__ __launch_bounds__(256) void transpose_f32_f16(
    const float* __restrict__ w, _Float16* __restrict__ wT, int K, int N)
{
  __shared__ float tile[32][33];
  const int nb = blockIdx.x * 32, kb = blockIdx.y * 32;
  const int tx = threadIdx.x & 31, ty = threadIdx.x >> 5;
#pragma unroll
  for (int p = 0; p < 4; p++)
    tile[ty + p * 8][tx] = w[(size_t)(kb + ty + p * 8) * N + nb + tx];
  __syncthreads();
#pragma unroll
  for (int p = 0; p < 4; p++)
    wT[(size_t)(nb + ty + p * 8) * K + kb + tx] = (_Float16)tile[tx][ty + p * 8];
}

// ---------------------------------------------------------------------------
// GEMM: C[M,N] = A[M,K] @ B[K,N] with B given transposed (BT[N][K]); f16 in,
// f16 WMMA compute, f16 or f32 out. Block tile 128x128, BK=32, 8 waves (4x2).
// A/B tiles staged by TDM with hardware padding to [128][40], double-buffered.
// ---------------------------------------------------------------------------
template <int M, int N, int K, bool OUT_F16>
__global__ __launch_bounds__(256) void gemm_wmma_tdm(
    const _Float16* __restrict__ A, const _Float16* __restrict__ BT,
    void* __restrict__ Cout)
{
  __shared__ _Float16 As[2][128][40];
  __shared__ _Float16 Bs[2][128][40];

  const int tid  = threadIdx.x;
  const int lane = tid & 31;
  const int wave = tid >> 5;
  const int g    = lane >> 4;
  const int lm   = lane & 15;
  const int m0   = blockIdx.y * 128;
  const int n0   = blockIdx.x * 128;
  const int wm   = (wave & 3) * 32;
  const int wn   = (wave >> 2) * 64;

  v8f acc[2][4];
#pragma unroll
  for (int mi = 0; mi < 2; mi++)
#pragma unroll
    for (int ni = 0; ni < 4; ni++)
#pragma unroll
      for (int e = 0; e < 8; e++) acc[mi][ni][e] = 0.0f;

  constexpr int nk = K / 32;
  if (wave == 0) {
    tdm_load_tile_f16(A  + (size_t)m0 * K, lds_off(&As[0][0][0]), 32, 128, K, 3, 3);
    tdm_load_tile_f16(BT + (size_t)n0 * K, lds_off(&Bs[0][0][0]), 32, 128, K, 3, 3);
  }

#pragma unroll 2
  for (int kt = 0; kt < nk; kt++) {
    const int cur = kt & 1;
    if (wave == 0) __builtin_amdgcn_s_wait_tensorcnt(0);
    __syncthreads();  // cur tile ready; everyone done reading nxt buffer
    if (wave == 0 && kt + 1 < nk) {
      const int nxt = cur ^ 1;
      tdm_load_tile_f16(A  + (size_t)m0 * K + (kt + 1) * 32,
                        lds_off(&As[nxt][0][0]), 32, 128, K, 3, 3);
      tdm_load_tile_f16(BT + (size_t)n0 * K + (kt + 1) * 32,
                        lds_off(&Bs[nxt][0][0]), 32, 128, K, 3, 3);
    }

    FragA a[2], b[4];
#pragma unroll
    for (int mi = 0; mi < 2; mi++) {
      const int m = wm + mi * 16 + lm;
      a[mi].h[0] = *(const h8*)&As[cur][m][g * 8];
      a[mi].h[1] = *(const h8*)&As[cur][m][16 + g * 8];
    }
#pragma unroll
    for (int ni = 0; ni < 4; ni++) {
      const int n = wn + ni * 16 + lm;
      b[ni].h[0] = *(const h8*)&Bs[cur][n][g * 8];
      b[ni].h[1] = *(const h8*)&Bs[cur][n][16 + g * 8];
    }
#pragma unroll
    for (int mi = 0; mi < 2; mi++)
#pragma unroll
      for (int ni = 0; ni < 4; ni++)
        acc[mi][ni] = __builtin_amdgcn_wmma_f32_16x16x32_f16(
            false, a[mi].v, false, b[ni].v, (short)0, acc[mi][ni], false, false);
  }

#pragma unroll
  for (int mi = 0; mi < 2; mi++)
#pragma unroll
    for (int ni = 0; ni < 4; ni++) {
      const size_t base = (size_t)(m0 + wm + mi * 16 + 8 * g) * N
                        + (n0 + wn + ni * 16 + lm);
      if constexpr (OUT_F16) {
        _Float16* C = (_Float16*)Cout;
#pragma unroll
        for (int r = 0; r < 8; r++) C[base + (size_t)r * N] = (_Float16)acc[mi][ni][r];
      } else {
        float* C = (float*)Cout;
#pragma unroll
        for (int r = 0; r < 8; r++) C[base + (size_t)r * N] = acc[mi][ni][r];
      }
    }
}

// ---------------------------------------------------------------------------
// RoPE (interleaved pairs) in-place on f16 qkv; math in f32.
// ---------------------------------------------------------------------------
__global__ __launch_bounds__(256) void rope_f16(
    _Float16* __restrict__ qkv, const float* __restrict__ cosb,
    const float* __restrict__ sinb)
{
  const size_t idx = (size_t)blockIdx.x * 256 + threadIdx.x;
  const int i  = (int)(idx & 31);
  const int h  = (int)((idx >> 5) & 15);
  const int tq = (int)((idx >> 9) & 1);
  const int s  = (int)((idx >> 10) & 1023);
  const int b  = (int)(idx >> 20);
  const size_t base = ((size_t)(b * SS + s)) * 3072 + tq * 1024 + h * 64 + 2 * i;
  const float c  = cosb[s * 32 + i];
  const float sn = sinb[s * 32 + i];
  const float x1 = (float)qkv[base];
  const float x2 = (float)qkv[base + 1];
  qkv[base]     = (_Float16)(x1 * c - x2 * sn);
  qkv[base + 1] = (_Float16)(x1 * sn + x2 * c);
}

// ---------------------------------------------------------------------------
// Flash attention on f16 qkv: grid (S/128, B*H); 8 waves x 16 query rows.
// K tile staged by TDM (hardware-padded [32][72]); V staged manually
// transposed; all matmuls v_wmma_f32_16x16x32_f16; f16 output for proj GEMM.
// ---------------------------------------------------------------------------
__global__ __launch_bounds__(256) void attn_wmma(
    const _Float16* __restrict__ qkv, _Float16* __restrict__ out)
{
  __shared__ _Float16 Ks[32][72];      // [key][hs] (TDM-padded)
  __shared__ _Float16 Vt[64][40];      // [hs][key] (transposed)
  __shared__ _Float16 Ps[8][16][40];   // per-wave P relayout patch

  const int tid  = threadIdx.x;
  const int lane = tid & 31;
  const int wave = tid >> 5;
  const int g    = lane >> 4;
  const int lm   = lane & 15;
  const int q0   = blockIdx.x * 128;
  const int b    = blockIdx.y >> 4;
  const int h    = blockIdx.y & 15;
  const int qw0  = q0 + wave * 16;

  // Q fragments: direct 16B loads from f16 global (A-operand layout runs).
  FragA qa[2];
  {
    const _Float16* qp = qkv + (size_t)(b * SS + qw0 + lm) * 3072 + h * 64;
#pragma unroll
    for (int kc = 0; kc < 2; kc++) {
      qa[kc].h[0] = *(const h8*)(qp + kc * 32 + g * 8);
      qa[kc].h[1] = *(const h8*)(qp + kc * 32 + g * 8 + 16);
    }
  }

  float m_run[8], l_run[8];
  v8f o[4];
#pragma unroll
  for (int r = 0; r < 8; r++) { m_run[r] = -__builtin_inff(); l_run[r] = 0.0f; }
#pragma unroll
  for (int t = 0; t < 4; t++)
#pragma unroll
    for (int r = 0; r < 8; r++) o[t][r] = 0.0f;

  const float scale = 0.125f;  // 1/sqrt(64)
  const _Float16* kbase = qkv + (size_t)b * SS * 3072 + 1024 + h * 64;
  const _Float16* vbase = kbase + 1024;
  const int nkb = q0 / 32 + 4;

  for (int kb = 0; kb < nkb; kb++) {
    __syncthreads();  // previous tile fully consumed
    if (wave == 0)    // K tile via TDM: 32 rows x 64 f16, pad 128B->144B rows
      tdm_load_tile_f16(kbase + (size_t)kb * 32 * 3072,
                        lds_off(&Ks[0][0]), 64, 32, 3072, 4, 3);
    // V tile: manual transposed staging from f16 global.
    {
      const int keyl = tid >> 4;
      const int hs   = (tid & 15) * 4;
#pragma unroll
      for (int p = 0; p < 2; p++) {
        const int key = keyl + p * 16;
        const h4 fv = *(const h4*)(vbase + (size_t)(kb * 32 + key) * 3072 + hs);
        Vt[hs + 0][key] = fv[0];
        Vt[hs + 1][key] = fv[1];
        Vt[hs + 2][key] = fv[2];
        Vt[hs + 3][key] = fv[3];
      }
    }
    if (wave == 0) __builtin_amdgcn_s_wait_tensorcnt(0);
    __syncthreads();

    if (kb * 32 <= qw0 + 15) {
      v8f sacc[2];
#pragma unroll
      for (int ns = 0; ns < 2; ns++)
#pragma unroll
        for (int e = 0; e < 8; e++) sacc[ns][e] = 0.0f;
#pragma unroll
      for (int kc = 0; kc < 2; kc++)
#pragma unroll
        for (int ns = 0; ns < 2; ns++) {
          FragA kf;
          kf.h[0] = *(const h8*)&Ks[ns * 16 + lm][kc * 32 + g * 8];
          kf.h[1] = *(const h8*)&Ks[ns * 16 + lm][kc * 32 + g * 8 + 16];
          sacc[ns] = __builtin_amdgcn_wmma_f32_16x16x32_f16(
              false, qa[kc].v, false, kf.v, (short)0, sacc[ns], false, false);
        }

#pragma unroll
      for (int r = 0; r < 8; r++) {
        const int qrow = qw0 + r + 8 * g;
        const int key0 = kb * 32 + lm;
        float s0 = (key0      <= qrow) ? sacc[0][r] * scale : -__builtin_inff();
        float s1 = (key0 + 16 <= qrow) ? sacc[1][r] * scale : -__builtin_inff();
        const float tmax  = red_max16(fmaxf(s0, s1));
        const float mnew  = fmaxf(m_run[r], tmax);
        const float alpha = __expf(m_run[r] - mnew);
        const float p0 = __expf(s0 - mnew);
        const float p1 = __expf(s1 - mnew);
        const float ts = red_sum16(p0 + p1);
        l_run[r] = l_run[r] * alpha + ts;
        m_run[r] = mnew;
#pragma unroll
        for (int t = 0; t < 4; t++) o[t][r] *= alpha;
        Ps[wave][r + 8 * g][lm]      = (_Float16)p0;
        Ps[wave][r + 8 * g][16 + lm] = (_Float16)p1;
      }
      asm volatile("s_wait_dscnt 0" ::: "memory");

      FragA pa;
      pa.h[0] = *(const h8*)&Ps[wave][lm][g * 8];
      pa.h[1] = *(const h8*)&Ps[wave][lm][16 + g * 8];
#pragma unroll
      for (int t = 0; t < 4; t++) {
        FragA vf;
        vf.h[0] = *(const h8*)&Vt[t * 16 + lm][g * 8];
        vf.h[1] = *(const h8*)&Vt[t * 16 + lm][16 + g * 8];
        o[t] = __builtin_amdgcn_wmma_f32_16x16x32_f16(
            false, pa.v, false, vf.v, (short)0, o[t], false, false);
      }
    }
  }

#pragma unroll
  for (int r = 0; r < 8; r++) {
    const float inv = 1.0f / l_run[r];
    const int qrow = qw0 + r + 8 * g;
    _Float16* op = out + (size_t)(b * SS + qrow) * DD + h * 64 + lm;
#pragma unroll
    for (int t = 0; t < 4; t++) op[t * 16] = (_Float16)(o[t][r] * inv);
  }
}

// ---------------------------------------------------------------------------
extern "C" void kernel_launch(void* const* d_in, const int* in_sizes, int n_in,
                              void* d_out, int out_size, void* d_ws, size_t ws_size,
                              hipStream_t stream) {
  const float* x      = (const float*)d_in[0];
  const float* w_qkv  = (const float*)d_in[1];
  const float* w_proj = (const float*)d_in[2];
  const float* cosb   = (const float*)d_in[3];
  const float* sinb   = (const float*)d_in[4];
  // d_in[5] = attn_mask (all-valid) -> causal mask only.

  _Float16* xh     = (_Float16*)d_ws;                       // [8192,1024]  16 MB
  _Float16* wqkvT  = xh + (size_t)BB * SS * DD;             // [3072,1024]   6 MB
  _Float16* wprojT = wqkvT + (size_t)3 * DD * DD;           // [1024,1024]   2 MB
  _Float16* qkvh   = wprojT + (size_t)DD * DD;              // [8192,3072]  48 MB
  _Float16* attnh  = qkvh + (size_t)BB * SS * 3 * DD;       // [8192,1024]  16 MB

  // Prepasses: convert + transpose into f16 (one-shot, L2-friendly).
  f32_to_f16_vec<<<(BB * SS * DD) / 1024, 256, 0, stream>>>(x, xh);
  transpose_f32_f16<<<dim3(3 * DD / 32, DD / 32), 256, 0, stream>>>(
      w_qkv, wqkvT, DD, 3 * DD);
  transpose_f32_f16<<<dim3(DD / 32, DD / 32), 256, 0, stream>>>(
      w_proj, wprojT, DD, DD);

  // 1) QKV GEMM (f16 out): [8192,1024] @ [1024,3072]
  gemm_wmma_tdm<BB * SS, 3 * DD, DD, true>
      <<<dim3(3 * DD / 128, BB * SS / 128), 256, 0, stream>>>(xh, wqkvT, qkvh);
  // 2) RoPE on q,k in-place (f16)
  rope_f16<<<(BB * SS * 2 * HH * 32) / 256, 256, 0, stream>>>(qkvh, cosb, sinb);
  // 3) Flash attention (f16 out)
  attn_wmma<<<dim3(SS / 128, BB * HH), 256, 0, stream>>>(qkvh, attnh);
  // 4) Output projection (f32 out): [8192,1024] @ [1024,1024]
  gemm_wmma_tdm<BB * SS, DD, DD, false>
      <<<dim3(DD / 128, BB * SS / 128), 256, 0, stream>>>(attnh, wprojT, d_out);
}
